// MultiHeadSelfAttention_39367670235392
// MI455X (gfx1250) — compile-verified, matrix operand order checked by reading, not run on hardware
//
#include <hip/hip_runtime.h>
#include <hip/hip_bf16.h>
#include <stdint.h>

// ---------------------------------------------------------------------------
// MI455X (gfx1250) fused multi-head self-attention, bf16 WMMA pipeline with
// double-buffered async global->LDS staging.
// B=4, S=2048, D=1024, H=16, Dk=64.
// ---------------------------------------------------------------------------

#define S_LEN   2048
#define DM      1024
#define NH      16
#define DK      64
#define NTOK    (4 * S_LEN)          // 8192 rows
#define KSTEP   64

typedef __attribute__((ext_vector_type(16))) __bf16 v16bf;
typedef __attribute__((ext_vector_type(8)))  float  v8f;
typedef __attribute__((ext_vector_type(2)))  int    v2i_t;

union Frag16 {
    v16bf    bf;
    uint32_t u[8];
};

// --- async global->LDS path (guarded; falls back to sync staging) ----------
#if __has_builtin(__builtin_amdgcn_global_load_async_to_lds_b64) && \
    __has_builtin(__builtin_amdgcn_s_wait_asynccnt)
#define HAVE_ASYNC_LDS 1
#else
#define HAVE_ASYNC_LDS 0
#endif

__device__ __forceinline__ void async_copy8(unsigned short* lds,
                                            const unsigned short* g) {
#if HAVE_ASYNC_LDS
    __builtin_amdgcn_global_load_async_to_lds_b64(
        (__attribute__((address_space(1))) v2i_t*)g,
        (__attribute__((address_space(3))) v2i_t*)lds,
        /*offset=*/0, /*cpol=*/0);
#else
    *(uint2*)lds = *(const uint2*)g;
#endif
}

__device__ __forceinline__ void async_fence_and_barrier() {
#if HAVE_ASYNC_LDS
    __builtin_amdgcn_s_wait_asynccnt(0);
#endif
    __syncthreads();
}

// float -> bf16, round to nearest even
__device__ __forceinline__ unsigned short f2bf(float f) {
    uint32_t u = __float_as_uint(f);
    uint32_t r = (u + 0x7FFFu + ((u >> 16) & 1u)) >> 16;
    return (unsigned short)r;
}

// Load a 16x32 bf16 WMMA fragment (A pattern; identical pattern serves B when
// the source is stored K-major per output column, i.e. B^T row-major).
// ISA 7.12.2: lane L -> row m=L&15, half h=L>>4; VGPR v0..3 = K 8h+2v(+1),
// VGPR v4..7 = K 16+8h+2(v-4)(+1)  -> two contiguous 16B chunks per lane.
__device__ __forceinline__ void load_frag(Frag16& f, const unsigned short* base,
                                          int ld, int lane) {
    const int m = lane & 15;
    const int h = (lane >> 4) & 1;
    const uint32_t* p0 = (const uint32_t*)(base + m * ld + 8 * h);
    const uint32_t* p1 = (const uint32_t*)(base + m * ld + 16 + 8 * h);
    f.u[0] = p0[0]; f.u[1] = p0[1]; f.u[2] = p0[2]; f.u[3] = p0[3];
    f.u[4] = p1[0]; f.u[5] = p1[1]; f.u[6] = p1[2]; f.u[7] = p1[3];
}

__device__ __forceinline__ v8f wmma_bf16(const Frag16& a, const Frag16& b, v8f c) {
    return __builtin_amdgcn_wmma_f32_16x16x32_bf16(
        /*neg_a=*/false, a.bf, /*neg_b=*/false, b.bf,
        /*c_mod=*/(short)0, c, /*reuse_a=*/false, /*reuse_b=*/false);
}

// ---------------------------------------------------------------------------
// Kernel 1: f32 -> bf16 elementwise convert (for x)
// ---------------------------------------------------------------------------
__global__ __launch_bounds__(256) void cvt_f32_to_bf16(const float* __restrict__ in,
                                                       unsigned short* __restrict__ out,
                                                       int n) {
    int i = blockIdx.x * 256 + threadIdx.x;
    if (i < n) out[i] = f2bf(in[i]);
}

// ---------------------------------------------------------------------------
// Kernel 2: weight transpose + convert: W[in][out] f32 -> Wt[out][in] bf16
// 32x32 LDS-tiled transpose, 256 threads (32x8)
// ---------------------------------------------------------------------------
__global__ __launch_bounds__(256) void transpose_cvt_w(const float* __restrict__ W,
                                                       unsigned short* __restrict__ Wt) {
    __shared__ __align__(16) float tile[32][33];
    const int bx = blockIdx.x * 32;   // out-col (n) block
    const int by = blockIdx.y * 32;   // in-row (k) block
    const int tx = threadIdx.x & 31;
    const int ty = threadIdx.x >> 5;  // 0..7
#pragma unroll
    for (int j = 0; j < 32; j += 8)
        tile[ty + j][tx] = W[(by + ty + j) * DM + bx + tx];
    __syncthreads();
#pragma unroll
    for (int j = 0; j < 32; j += 8)
        Wt[(bx + ty + j) * DM + by + tx] = f2bf(tile[tx][ty + j]);
}

// ---------------------------------------------------------------------------
// Kernel 3: bf16 GEMM  C[M][N] = A[M][K] * Bt[N][K]^T + bias
// Block tile 128x128, 256 threads = 8 waves (4 along M x 2 along N),
// wave tile 32x64 (2x4 accumulators), k-step 64, double-buffered async LDS.
// 16 WMMAs per barrier.
// ---------------------------------------------------------------------------
template <bool OUT_F32>
__global__ __launch_bounds__(256) void gemm_bf16_wmma(
    const unsigned short* __restrict__ A,   // [M][K]
    const unsigned short* __restrict__ Bt,  // [N][K]
    const float* __restrict__ bias,         // [N]
    void* __restrict__ Out,                 // [M][N] f32 or bf16
    int Mdim, int Ndim, int Kdim) {
    __shared__ __align__(16) unsigned short As[2][128 * KSTEP];
    __shared__ __align__(16) unsigned short Bs[2][128 * KSTEP];

    const int tid  = threadIdx.x;
    const int wave = tid >> 5;
    const int lane = tid & 31;
    const int m    = lane & 15;
    const int hlf  = (lane >> 4) & 1;
    const int wm   = wave & 3;   // 0..3  -> M offset wm*32
    const int wn   = wave >> 2;  // 0..1  -> N offset wn*64
    const int m0   = blockIdx.x * 128;
    const int n0   = blockIdx.y * 128;

    v8f acc[2][4] = {};

    // stage one 128xKSTEP tile of A and Bt into LDS buffer `buf`
    auto stage = [&](int buf, int k0) {
#pragma unroll
        for (int i = 0; i < 8; i++) {
            const int c   = tid + 256 * i;   // 0..2047 chunk id (4 bf16 each)
            const int row = c >> 4;          // 16 chunks per 64-elem row
            const int kk  = (c & 15) * 4;
            async_copy8(&As[buf][row * KSTEP + kk],
                        &A[(long)(m0 + row) * Kdim + k0 + kk]);
            async_copy8(&Bs[buf][row * KSTEP + kk],
                        &Bt[(long)(n0 + row) * Kdim + k0 + kk]);
        }
    };

    stage(0, 0);
    async_fence_and_barrier();

    int cur = 0;
    for (int k0 = 0; k0 < Kdim; k0 += KSTEP) {
        // kick off next tile into the other buffer while we compute
        if (k0 + KSTEP < Kdim) stage(cur ^ 1, k0 + KSTEP);

        Frag16 af[2][2], bf[4][2];
#pragma unroll
        for (int mt = 0; mt < 2; mt++)
#pragma unroll
            for (int kf = 0; kf < 2; kf++)
                load_frag(af[mt][kf],
                          &As[cur][(wm * 32 + mt * 16) * KSTEP + kf * 32],
                          KSTEP, lane);
#pragma unroll
        for (int nt = 0; nt < 4; nt++)
#pragma unroll
            for (int kf = 0; kf < 2; kf++)
                load_frag(bf[nt][kf],
                          &Bs[cur][(wn * 64 + nt * 16) * KSTEP + kf * 32],
                          KSTEP, lane);
#pragma unroll
        for (int mt = 0; mt < 2; mt++)
#pragma unroll
            for (int nt = 0; nt < 4; nt++)
#pragma unroll
                for (int kf = 0; kf < 2; kf++)
                    acc[mt][nt] = wmma_bf16(af[mt][kf], bf[nt][kf], acc[mt][nt]);

        async_fence_and_barrier();
        cur ^= 1;
    }

#pragma unroll
    for (int mt = 0; mt < 2; mt++)
#pragma unroll
        for (int nt = 0; nt < 4; nt++)
#pragma unroll
            for (int r = 0; r < 8; r++) {
                const int gm = m0 + wm * 32 + mt * 16 + r + 8 * hlf;
                const int gn = n0 + wn * 64 + nt * 16 + m;
                const float v = acc[mt][nt][r] + bias[gn];
                if (OUT_F32)
                    ((float*)Out)[(long)gm * Ndim + gn] = v;
                else
                    ((unsigned short*)Out)[(long)gm * Ndim + gn] = f2bf(v);
            }
}

// ---------------------------------------------------------------------------
// Kernel 4: flash attention, bf16 WMMA, online softmax.
// grid = (S/128, B*H); block = 256 (8 waves). Wave w owns Q rows w*16..+15.
// 64 keys/iter; double-buffered K (async) and V^T (manual transpose) tiles.
// ---------------------------------------------------------------------------
__global__ __launch_bounds__(256) void attn_flash_wmma(
    const unsigned short* __restrict__ Qb,   // [NTOK][DM] bf16
    const unsigned short* __restrict__ Kb,
    const unsigned short* __restrict__ Vb,
    unsigned short* __restrict__ Ctx) {      // [NTOK][DM] bf16
    __shared__ __align__(16) unsigned short ldsK[2][64 * 64];
    __shared__ __align__(16) unsigned short ldsVT[2][64 * 64];
    __shared__ __align__(16) unsigned short ldsP[8][16 * 64];

    const int tid  = threadIdx.x;
    const int wave = tid >> 5;
    const int lane = tid & 31;
    const int m    = lane & 15;
    const int hlf  = (lane >> 4) & 1;

    const int bh   = blockIdx.y;          // b*16 + h
    const int b    = bh >> 4;
    const int h    = bh & 15;
    const int dcol = h * DK;
    const long rowbase = (long)b * S_LEN + blockIdx.x * 128;

    // Q fragments resident in registers: 16 rows x 64 (two K=32 slices)
    Frag16 qf0, qf1;
    {
        const unsigned short* qbase = Qb + (rowbase + wave * 16) * DM + dcol;
        load_frag(qf0, qbase, DM, lane);
        load_frag(qf1, qbase + 32, DM, lane);
    }

    v8f acc[4] = {};
    float mprev[8], lsum[8];
#pragma unroll
    for (int r = 0; r < 8; r++) { mprev[r] = -1e30f; lsum[r] = 0.f; }

    const float scale = 0.125f;  // 1/sqrt(64)

    // stage one 64-key tile: K row-major via async copies, V transposed
    auto stageKV = [&](int buf, int t0) {
        const long tilebase = ((long)b * S_LEN + t0) * DM + dcol;
#pragma unroll
        for (int i = 0; i < 4; i++) {
            const int c  = tid + 256 * i;   // 0..1023 chunks of 4 bf16
            const int tt = c >> 4;          // 16 chunks per 64-elem row
            const int kk = (c & 15) * 4;
            async_copy8(&ldsK[buf][tt * 64 + kk],
                        &Kb[tilebase + (long)tt * DM + kk]);
        }
#pragma unroll
        for (int e = 0; e < 16; e++) {
            const int lin = tid + 256 * e;  // 0..4095
            const int tt  = lin >> 6;
            const int dd  = lin & 63;
            ldsVT[buf][dd * 64 + tt] = Vb[tilebase + (long)tt * DM + dd];
        }
    };

    stageKV(0, 0);
    async_fence_and_barrier();

    int cur = 0;
    for (int t0 = 0; t0 < S_LEN; t0 += 64) {
        if (t0 + 64 < S_LEN) {
            stageKV(cur ^ 1, t0 + 64);
            // pull the tile after next into L2 under the WMMA work
            if (t0 + 128 < S_LEN) {
                const long nrow = (long)b * S_LEN + t0 + 128 + (tid >> 2);
                __builtin_prefetch(&Kb[nrow * DM + dcol], 0, 0);
                __builtin_prefetch(&Vb[nrow * DM + dcol], 0, 0);
            }
        }

        // S = Q K^T  (4 tiles of 16 cols; K rows ARE B^T for QK^T)
        v8f s[4];
#pragma unroll
        for (int nt = 0; nt < 4; nt++) {
            Frag16 kf0, kf1;
            load_frag(kf0, &ldsK[cur][(nt * 16) * 64], 64, lane);
            load_frag(kf1, &ldsK[cur][(nt * 16) * 64 + 32], 64, lane);
            v8f z = {};
            z = wmma_bf16(qf0, kf0, z);
            z = wmma_bf16(qf1, kf1, z);
            s[nt] = z;
        }
#pragma unroll
        for (int nt = 0; nt < 4; nt++)
#pragma unroll
            for (int r = 0; r < 8; r++) s[nt][r] *= scale;

        // online softmax. C layout: row = r + 8*hlf, col = (lane&15) + 16*nt.
        // Row reduction = xor-shuffle over lane bits 0..3 (16-lane halves).
        float mnew[8], corr[8];
#pragma unroll
        for (int r = 0; r < 8; r++) {
            float v = fmaxf(fmaxf(s[0][r], s[1][r]), fmaxf(s[2][r], s[3][r]));
            v = fmaxf(v, __shfl_xor(v, 1, 32));
            v = fmaxf(v, __shfl_xor(v, 2, 32));
            v = fmaxf(v, __shfl_xor(v, 4, 32));
            v = fmaxf(v, __shfl_xor(v, 8, 32));
            mnew[r] = fmaxf(mprev[r], v);
            corr[r] = __expf(mprev[r] - mnew[r]);
        }

        unsigned short* Pw = &ldsP[wave][0];
        float rsum[8];
#pragma unroll
        for (int r = 0; r < 8; r++) rsum[r] = 0.f;
#pragma unroll
        for (int nt = 0; nt < 4; nt++)
#pragma unroll
            for (int r = 0; r < 8; r++) {
                const float p = __expf(s[nt][r] - mnew[r]);
                rsum[r] += p;
                Pw[(r + 8 * hlf) * 64 + nt * 16 + m] = f2bf(p);
            }
#pragma unroll
        for (int r = 0; r < 8; r++) {
            float rs = rsum[r];
            rs += __shfl_xor(rs, 1, 32);
            rs += __shfl_xor(rs, 2, 32);
            rs += __shfl_xor(rs, 4, 32);
            rs += __shfl_xor(rs, 8, 32);
            lsum[r]  = lsum[r] * corr[r] + rs;
            mprev[r] = mnew[r];
#pragma unroll
            for (int nt = 0; nt < 4; nt++) acc[nt][r] *= corr[r];
        }

        // ctx += P V  (P A-fragments reloaded from LDS; V^T rows are B^T)
        Frag16 pf0, pf1;
        load_frag(pf0, Pw, 64, lane);
        load_frag(pf1, Pw + 32, 64, lane);
#pragma unroll
        for (int nt = 0; nt < 4; nt++) {
            Frag16 vf0, vf1;
            load_frag(vf0, &ldsVT[cur][(nt * 16) * 64], 64, lane);
            load_frag(vf1, &ldsVT[cur][(nt * 16) * 64 + 32], 64, lane);
            acc[nt] = wmma_bf16(pf0, vf0, acc[nt]);
            acc[nt] = wmma_bf16(pf1, vf1, acc[nt]);
        }

        async_fence_and_barrier();
        cur ^= 1;
    }

    // finalize: ctx / rowsum -> bf16 workspace
#pragma unroll
    for (int nt = 0; nt < 4; nt++)
#pragma unroll
        for (int r = 0; r < 8; r++) {
            const long grow = rowbase + wave * 16 + r + 8 * hlf;
            const float v = acc[nt][r] / lsum[r];
            Ctx[grow * DM + dcol + nt * 16 + m] = f2bf(v);
        }
}

// ---------------------------------------------------------------------------
// Launch
// ---------------------------------------------------------------------------
extern "C" void kernel_launch(void* const* d_in, const int* in_sizes, int n_in,
                              void* d_out, int out_size, void* d_ws, size_t ws_size,
                              hipStream_t stream) {
    const float* x   = (const float*)d_in[0];
    const float* w_q = (const float*)d_in[1];
    const float* b_q = (const float*)d_in[2];
    const float* w_k = (const float*)d_in[3];
    const float* b_k = (const float*)d_in[4];
    const float* w_v = (const float*)d_in[5];
    const float* b_v = (const float*)d_in[6];
    const float* w_o = (const float*)d_in[7];
    const float* b_o = (const float*)d_in[8];
    float* out = (float*)d_out;

    // workspace carve-up (bf16 elements): 44M elems = 88 MB
    unsigned short* ws  = (unsigned short*)d_ws;
    const long MT = (long)NTOK * DM;   // 8M
    const long WT = (long)DM * DM;     // 1M
    unsigned short* xb   = ws;
    unsigned short* wqt  = xb + MT;
    unsigned short* wkt  = wqt + WT;
    unsigned short* wvt  = wkt + WT;
    unsigned short* wot  = wvt + WT;
    unsigned short* qb   = wot + WT;
    unsigned short* kb   = qb + MT;
    unsigned short* vb   = kb + MT;
    unsigned short* ctxb = vb + MT;

    // 1) convert activations, transpose+convert weights
    cvt_f32_to_bf16<<<(int)(MT / 256), 256, 0, stream>>>(x, xb, (int)MT);
    dim3 tg(DM / 32, DM / 32);
    transpose_cvt_w<<<tg, 256, 0, stream>>>(w_q, wqt);
    transpose_cvt_w<<<tg, 256, 0, stream>>>(w_k, wkt);
    transpose_cvt_w<<<tg, 256, 0, stream>>>(w_v, wvt);
    transpose_cvt_w<<<tg, 256, 0, stream>>>(w_o, wot);

    // 2) QKV projections (bf16 out)
    dim3 gg(NTOK / 128, DM / 128);   // 64 x 8
    gemm_bf16_wmma<false><<<gg, 256, 0, stream>>>(xb, wqt, b_q, qb, NTOK, DM, DM);
    gemm_bf16_wmma<false><<<gg, 256, 0, stream>>>(xb, wkt, b_k, kb, NTOK, DM, DM);
    gemm_bf16_wmma<false><<<gg, 256, 0, stream>>>(xb, wvt, b_v, vb, NTOK, DM, DM);

    // 3) flash attention
    dim3 ga(S_LEN / 128, 4 * NH);    // 16 x 64
    attn_flash_wmma<<<ga, 256, 0, stream>>>(qb, kb, vb, ctxb);

    // 4) output projection (f32 out)
    gemm_bf16_wmma<true><<<gg, 256, 0, stream>>>(ctxb, wot, b_o, out, NTOK, DM, DM);
}
